// Model_19859928777324
// MI455X (gfx1250) — compile-verified, matrix operand order checked by reading, not run on hardware
//
#include <hip/hip_runtime.h>
#include <hip/hip_bf16.h>
#include <math.h>

// ---------------- model dims ----------------
#define BB       8
#define LL       512
#define ENC_IN   7
#define MARK     4
#define D_MODEL  512
#define D_INNER  1024
#define D_STATE  16
#define D_CONV   4
#define DT_RANK  32
#define PRED_LEN 96
#define C_OUT    7
#define NTOK     (BB*LL)            // 4096 rows

typedef __attribute__((ext_vector_type(16))) __bf16 v16bf;
typedef __attribute__((ext_vector_type(8)))  float  v8f;
typedef __attribute__((ext_vector_type(4)))  int    v4i;

// ---------------- CDNA5 async global->LDS copy (guarded) ----------------
#if defined(__HIP_DEVICE_COMPILE__) && defined(__gfx1250__) && \
    __has_builtin(__builtin_amdgcn_global_load_async_to_lds_b128) && \
    __has_builtin(__builtin_amdgcn_s_wait_asynccnt)
#define USE_ASYNC_LDS 1
#else
#define USE_ASYNC_LDS 0
#endif

__device__ __forceinline__ void cp16_g2l(__bf16* dst_lds, const __bf16* src_glob) {
#if USE_ASYNC_LDS
    // builtin takes generic int4* pointers (per compiler diagnostic); backend
    // resolves global vs LDS address spaces.
    __builtin_amdgcn_global_load_async_to_lds_b128(
        (v4i*)(uintptr_t)src_glob, (v4i*)(uintptr_t)dst_lds, 0, 0);
#else
    *(uint4*)dst_lds = *(const uint4*)src_glob;
#endif
}

__device__ __forceinline__ void cp_wait_all() {
#if USE_ASYNC_LDS
    __builtin_amdgcn_s_wait_asynccnt(0);
#endif
}

// ---------------- workspace layout (bytes, 256-aligned) ----------------
#define OFF_MEAN   ((size_t)0)                      // 56 f32
#define OFF_STD    ((size_t)256)                    // 56 f32
#define OFF_XBF    ((size_t)512)                    // 4096x512 bf16
#define OFF_WA     (OFF_XBF   + (size_t)NTOK*D_MODEL*2)       // 1024x512 bf16
#define OFF_WXP    (OFF_WA    + (size_t)D_INNER*D_MODEL*2)    // 64x1024 bf16
#define OFF_WDT    (OFF_WXP   + (size_t)64*D_INNER*2)         // 1024x32 bf16
#define OFF_XPART  (OFF_WDT   + (size_t)D_INNER*DT_RANK*2)    // 4096x1024 f32
#define OFF_U      (OFF_XPART + (size_t)NTOK*D_INNER*4)       // 4096x1024 f32
#define OFF_UBF    (OFF_U     + (size_t)NTOK*D_INNER*4)       // 4096x1024 bf16
#define OFF_DBL    (OFF_UBF   + (size_t)NTOK*D_INNER*2)       // 4096x64 f32
#define OFF_DTLOW  (OFF_DBL   + (size_t)NTOK*64*4)            // 4096x32 bf16
#define OFF_DTRAW  (OFF_DTLOW + (size_t)NTOK*DT_RANK*2)       // 4096x1024 f32
#define OFF_ZL     (OFF_DTRAW + (size_t)NTOK*D_INNER*4)       // 8x1024 f32
#define OFF_YL     (OFF_ZL    + (size_t)BB*D_INNER*4)         // 8x1024 f32
#define OFF_XO     (OFF_YL    + (size_t)BB*D_INNER*4)         // 8x512 f32

__device__ __forceinline__ float siluf(float x) { return x / (1.0f + __expf(-x)); }

// ---- per-(b,c) mean / std over L ----
__global__ void k_stats(const float* __restrict__ x_enc, float* __restrict__ mean,
                        float* __restrict__ stdv) {
    int bc = blockIdx.x;                 // 0..55
    int b = bc / ENC_IN, c = bc % ENC_IN;
    int t = threadIdx.x;                 // 256
    float s = 0.f, ss = 0.f;
    for (int l = t; l < LL; l += 256) {
        float v = x_enc[((size_t)b*LL + l)*ENC_IN + c];
        s += v; ss += v*v;
    }
    __shared__ float sh[256], sh2[256];
    sh[t] = s; sh2[t] = ss; __syncthreads();
    for (int o = 128; o > 0; o >>= 1) {
        if (t < o) { sh[t] += sh[t+o]; sh2[t] += sh2[t+o]; }
        __syncthreads();
    }
    if (t == 0) {
        float m = sh[0] / (float)LL;
        float var = sh2[0] / (float)LL - m*m;
        mean[bc] = m;
        stdv[bc] = sqrtf(var + 1e-5f);
    }
}

// ---- f32 -> bf16 conversion ----
__global__ void k_cvt(const float* __restrict__ in, __bf16* __restrict__ out, int n) {
    int g = blockIdx.x*256 + threadIdx.x;
    if (g < n) out[g] = (__bf16)in[g];
}

// ---- embedding: circular token conv + temporal proj + positional, output bf16 ----
__global__ void k_embed(const float* __restrict__ x_enc, const float* __restrict__ xm,
                        const float* __restrict__ tok_w, const float* __restrict__ temp_w,
                        const float* __restrict__ mean, const float* __restrict__ stdv,
                        __bf16* __restrict__ xbf) {
    int g = blockIdx.x*256 + threadIdx.x;          // over 4096*512
    if (g >= NTOK*D_MODEL) return;
    int d  = g % D_MODEL;
    int bl = g / D_MODEL;
    int l  = bl % LL, b = bl / LL;
    float acc = 0.f;
    #pragma unroll
    for (int k = 0; k < 3; ++k) {
        int lp = l + k - 1; lp = (lp + LL) % LL;
        #pragma unroll
        for (int c = 0; c < ENC_IN; ++c) {
            float xn = (x_enc[((size_t)b*LL + lp)*ENC_IN + c] - mean[b*ENC_IN + c])
                       / stdv[b*ENC_IN + c];
            acc += xn * tok_w[((size_t)d*ENC_IN + c)*3 + k];
        }
    }
    #pragma unroll
    for (int m = 0; m < MARK; ++m)
        acc += xm[((size_t)b*LL + l)*MARK + m] * temp_w[(size_t)d*MARK + m];
    // positional embedding
    int i = d >> 1;
    float div = __expf((float)(2*i) * (-logf(10000.f) / (float)D_MODEL));
    float ang = (float)l * div;
    acc += (d & 1) ? __cosf(ang) : __sinf(ang);
    xbf[g] = (__bf16)acc;
}

// ---- bf16 WMMA GEMM: C[M,N] = A[M,K] * W[N,K]^T ----
// Block tile (4*WM) x (2*WN), 8 wave32s each computing WM x WN via 16x16x32 bf16 WMMA.
// Double-buffered LDS, async global->LDS staging when available.
template<int WM, int WN>
__global__ void k_gemm(const __bf16* __restrict__ A, const __bf16* __restrict__ W,
                       float* __restrict__ C, int M, int N, int K) {
    constexpr int BM = 4*WM, BN = 2*WN;
    constexpr int MI = WM/16, NI = WN/16;
    __shared__ __align__(64) __bf16 As[2][BM][32];
    __shared__ __align__(64) __bf16 Ws[2][BN][32];
    int tid  = threadIdx.x;                 // 256 = 8 wave32
    int wv   = tid >> 5;
    int lane = tid & 31;
    int r16  = lane & 15, hf = lane >> 4;
    int wm = wv & 3;                        // 4 wave tiles along M
    int wn = wv >> 2;                       // 2 wave tiles along N
    int m0 = blockIdx.y * BM;
    int n0 = blockIdx.x * BN;
    v8f acc[MI][NI] = {};
    int ktiles = K >> 5;

    auto stage = [&](int kt, int buf) {
        int k0 = kt << 5;
        #pragma unroll
        for (int i = 0; i < BM/64; ++i) {           // BM*32 bf16 = BM*4 16B-chunks
            int ch = tid + i*256;
            int r = ch >> 2, c8 = (ch & 3)*8;
            cp16_g2l(&As[buf][r][c8], &A[(size_t)(m0 + r)*K + k0 + c8]);
        }
        #pragma unroll
        for (int i = 0; i < BN/64; ++i) {
            int ch = tid + i*256;
            int r = ch >> 2, c8 = (ch & 3)*8;
            cp16_g2l(&Ws[buf][r][c8], &W[(size_t)(n0 + r)*K + k0 + c8]);
        }
    };

    stage(0, 0);
    cp_wait_all();
    __syncthreads();
    for (int kt = 0; kt < ktiles; ++kt) {
        int cur = kt & 1;
        if (kt + 1 < ktiles) stage(kt + 1, cur ^ 1);   // overlap with compute below
        v16bf a[MI], bfr[NI];
        #pragma unroll
        for (int mi = 0; mi < MI; ++mi)
            a[mi] = *(const v16bf*)&As[cur][wm*WM + mi*16 + r16][hf*16];
        #pragma unroll
        for (int ni = 0; ni < NI; ++ni)
            bfr[ni] = *(const v16bf*)&Ws[cur][wn*WN + ni*16 + r16][hf*16];
        #pragma unroll
        for (int mi = 0; mi < MI; ++mi)
            #pragma unroll
            for (int ni = 0; ni < NI; ++ni)
                acc[mi][ni] = __builtin_amdgcn_wmma_f32_16x16x32_bf16(
                    false, a[mi], false, bfr[ni], (short)0, acc[mi][ni], false, false);
        cp_wait_all();
        __syncthreads();
    }
    // C/D layout: lanes 0-15 -> M=v, lanes 16-31 -> M=v+8; N = lane&15
    #pragma unroll
    for (int mi = 0; mi < MI; ++mi) {
        float* Cp = C + (size_t)(m0 + wm*WM + mi*16)*N + n0 + wn*WN;
        #pragma unroll
        for (int ni = 0; ni < NI; ++ni)
            #pragma unroll
            for (int v = 0; v < 8; ++v)
                Cp[(size_t)(v + 8*hf)*N + ni*16 + r16] = acc[mi][ni][v];
    }
}

// ---- causal depthwise conv (k=4) + SiLU -> u (f32 + bf16) ----
__global__ void k_dwconv(const float* __restrict__ xpart, const float* __restrict__ cw,
                         const float* __restrict__ cb, float* __restrict__ u,
                         __bf16* __restrict__ ubf) {
    int g = blockIdx.x*256 + threadIdx.x;     // over 4096*1024
    if (g >= NTOK*D_INNER) return;
    int d  = g % D_INNER;
    int bl = g / D_INNER;
    int l  = bl % LL, b = bl / LL;
    float acc = cb[d];
    #pragma unroll
    for (int k = 0; k < D_CONV; ++k) {
        int lp = l + k - (D_CONV - 1);
        if (lp >= 0)
            acc += xpart[((size_t)b*LL + lp)*D_INNER + d] * cw[d*D_CONV + k];
    }
    float s = siluf(acc);
    u[g]   = s;
    ubf[g] = (__bf16)s;
}

// ---- split dbl -> dt_low (bf16, K=32 GEMM input) ----
__global__ void k_splitdt(const float* __restrict__ dbl, __bf16* __restrict__ dtlow) {
    int g = blockIdx.x*256 + threadIdx.x;     // over 4096*32
    if (g >= NTOK*DT_RANK) return;
    int j = g % DT_RANK, r = g / DT_RANK;
    dtlow[g] = (__bf16)dbl[(size_t)r*64 + j];
}

// ---- z at last timestep only: z[b,d] = x[b,L-1,:] . in_proj_w[1024+d,:] ----
__global__ void k_zlast(const __bf16* __restrict__ xbf, const float* __restrict__ wz,
                        float* __restrict__ zl) {
    int g = blockIdx.x*256 + threadIdx.x;     // 8*1024
    if (g >= BB*D_INNER) return;
    int d = g % D_INNER, b = g / D_INNER;
    const __bf16* xr = xbf + ((size_t)b*LL + (LL-1))*D_MODEL;
    const float*  wr = wz + (size_t)d*D_MODEL;
    float acc = 0.f;
    for (int k = 0; k < D_MODEL; ++k) acc += (float)xr[k] * wr[k];
    zl[g] = acc;
}

// ---- selective scan: one thread per (b,d), 16-state recurrence in registers.
//      B broadcast is chunked: 64 timesteps of B preloaded per barrier pair. ----
#define BCH 64
__global__ void k_scan(const float* __restrict__ dtraw, const float* __restrict__ dtb,
                       const float* __restrict__ u, const float* __restrict__ dbl,
                       const float* __restrict__ A_log, const float* __restrict__ Dp,
                       const float* __restrict__ zl, float* __restrict__ yl) {
    int b = blockIdx.x >> 2;                               // 4 blocks per batch
    int d = (blockIdx.x & 3)*256 + threadIdx.x;            // 0..1023
    float Ae[D_STATE], h[D_STATE];
    #pragma unroll
    for (int n = 0; n < D_STATE; ++n) {
        Ae[n] = -__expf(A_log[(size_t)d*D_STATE + n]);
        h[n] = 0.f;
    }
    float bias = dtb[d];
    __shared__ float Bsh[BCH][D_STATE];
    float uv = 0.f;
    for (int lc = 0; lc < LL; lc += BCH) {
        // cooperative preload of B rows for BCH timesteps
        for (int i = threadIdx.x; i < BCH*D_STATE; i += 256) {
            int t = i >> 4, n = i & 15;
            Bsh[t][n] = dbl[((size_t)b*LL + lc + t)*64 + DT_RANK + n];
        }
        __syncthreads();
        for (int t = 0; t < BCH; ++t) {
            size_t row = (size_t)b*LL + lc + t;
            float x = dtraw[row*D_INNER + d] + bias;
            float dt = (x > 20.f) ? x : log1pf(__expf(x));      // softplus
            uv = u[row*D_INNER + d];
            float dtu = dt * uv;
            #pragma unroll
            for (int n = 0; n < D_STATE; ++n)
                h[n] = __expf(dt * Ae[n]) * h[n] + dtu * Bsh[t][n];
        }
        __syncthreads();
    }
    // y at last step only: y = h.C_last + u_last*D, gated by silu(z_last)
    size_t lrow = (size_t)b*LL + (LL-1);
    float y = uv * Dp[d];
    #pragma unroll
    for (int n = 0; n < D_STATE; ++n)
        y += h[n] * dbl[lrow*64 + DT_RANK + D_STATE + n];
    y *= siluf(zl[(size_t)b*D_INNER + d]);
    yl[(size_t)b*D_INNER + d] = y;
}

// ---- out_proj at last timestep: (8x1024)(1024x512) ----
__global__ void k_outproj(const float* __restrict__ yl, const float* __restrict__ w,
                          float* __restrict__ xo) {
    int g = blockIdx.x*256 + threadIdx.x;     // 8*512
    if (g >= BB*D_MODEL) return;
    int j = g % D_MODEL, b = g / D_MODEL;
    const float* yr = yl + (size_t)b*D_INNER;
    const float* wr = w  + (size_t)j*D_INNER;
    float acc = 0.f;
    for (int k = 0; k < D_INNER; ++k) acc += yr[k] * wr[k];
    xo[g] = acc;
}

// ---- head + de-normalization ----
__global__ void k_head(const float* __restrict__ xo, const float* __restrict__ hw,
                       const float* __restrict__ hb, const float* __restrict__ stdv,
                       const float* __restrict__ mean, float* __restrict__ out) {
    int g = blockIdx.x*256 + threadIdx.x;     // 8*672
    if (g >= BB*PRED_LEN*C_OUT) return;
    int j = g % (PRED_LEN*C_OUT), b = g / (PRED_LEN*C_OUT);
    int p = j / C_OUT, c = j % C_OUT;
    const float* xr = xo + (size_t)b*D_MODEL;
    const float* wr = hw + (size_t)j*D_MODEL;
    float acc = hb[j];
    for (int k = 0; k < D_MODEL; ++k) acc += xr[k] * wr[k];
    out[((size_t)b*PRED_LEN + p)*C_OUT + c] = acc * stdv[b*C_OUT + c] + mean[b*C_OUT + c];
}

extern "C" void kernel_launch(void* const* d_in, const int* in_sizes, int n_in,
                              void* d_out, int out_size, void* d_ws, size_t ws_size,
                              hipStream_t stream) {
    const float* x_enc   = (const float*)d_in[0];
    const float* x_mark  = (const float*)d_in[1];
    const float* tok_w   = (const float*)d_in[4];
    const float* temp_w  = (const float*)d_in[5];
    const float* inproj  = (const float*)d_in[6];
    const float* conv_w  = (const float*)d_in[7];
    const float* conv_b  = (const float*)d_in[8];
    const float* xproj_w = (const float*)d_in[9];
    const float* dtproj_w= (const float*)d_in[10];
    const float* dtproj_b= (const float*)d_in[11];
    const float* A_log   = (const float*)d_in[12];
    const float* D_param = (const float*)d_in[13];
    const float* outp_w  = (const float*)d_in[14];
    const float* head_w  = (const float*)d_in[15];
    const float* head_b  = (const float*)d_in[16];
    float* out = (float*)d_out;

    char* ws = (char*)d_ws;
    float*  mean   = (float*)(ws + OFF_MEAN);
    float*  stdv   = (float*)(ws + OFF_STD);
    __bf16* xbf    = (__bf16*)(ws + OFF_XBF);
    __bf16* wA     = (__bf16*)(ws + OFF_WA);
    __bf16* wXp    = (__bf16*)(ws + OFF_WXP);
    __bf16* wDt    = (__bf16*)(ws + OFF_WDT);
    float*  xpart  = (float*)(ws + OFF_XPART);
    float*  u      = (float*)(ws + OFF_U);
    __bf16* ubf    = (__bf16*)(ws + OFF_UBF);
    float*  dbl    = (float*)(ws + OFF_DBL);
    __bf16* dtlow  = (__bf16*)(ws + OFF_DTLOW);
    float*  dtraw  = (float*)(ws + OFF_DTRAW);
    float*  zl     = (float*)(ws + OFF_ZL);
    float*  yl     = (float*)(ws + OFF_YL);
    float*  xo     = (float*)(ws + OFF_XO);

    // stats
    k_stats<<<BB*ENC_IN, 256, 0, stream>>>(x_enc, mean, stdv);
    // weight conversions to bf16
    k_cvt<<<(D_INNER*D_MODEL + 255)/256, 256, 0, stream>>>(inproj, wA, D_INNER*D_MODEL);
    k_cvt<<<(64*D_INNER + 255)/256, 256, 0, stream>>>(xproj_w, wXp, 64*D_INNER);
    k_cvt<<<(D_INNER*DT_RANK + 255)/256, 256, 0, stream>>>(dtproj_w, wDt, D_INNER*DT_RANK);
    // embedding
    k_embed<<<(NTOK*D_MODEL + 255)/256, 256, 0, stream>>>(x_enc, x_mark, tok_w, temp_w,
                                                          mean, stdv, xbf);
    // in_proj (x half): 4096x1024x512, 128x128 block tiles
    { dim3 g(D_INNER/128, NTOK/128);
      k_gemm<32,64><<<g, 256, 0, stream>>>(xbf, wA, xpart, NTOK, D_INNER, D_MODEL); }
    // depthwise conv + silu
    k_dwconv<<<(NTOK*D_INNER + 255)/256, 256, 0, stream>>>(xpart, conv_w, conv_b, u, ubf);
    // x_proj: 4096x64x1024, 64x64 block tiles (N=64)
    { dim3 g(64/64, NTOK/64);
      k_gemm<16,32><<<g, 256, 0, stream>>>(ubf, wXp, dbl, NTOK, 64, D_INNER); }
    // dt split to bf16
    k_splitdt<<<(NTOK*DT_RANK + 255)/256, 256, 0, stream>>>(dbl, dtlow);
    // dt_proj: 4096x1024x32, 128x128 block tiles, single K-step
    { dim3 g(D_INNER/128, NTOK/128);
      k_gemm<32,64><<<g, 256, 0, stream>>>(dtlow, wDt, dtraw, NTOK, D_INNER, DT_RANK); }
    // z at last timestep (second half of in_proj weights)
    k_zlast<<<(BB*D_INNER + 255)/256, 256, 0, stream>>>(xbf, inproj + (size_t)D_INNER*D_MODEL, zl);
    // selective scan (sequential over L, parallel over 8192 channels)
    k_scan<<<BB*4, 256, 0, stream>>>(dtraw, dtproj_b, u, dbl, A_log, D_param, zl, yl);
    // out_proj + head + denorm (last timestep only)
    k_outproj<<<(BB*D_MODEL + 255)/256, 256, 0, stream>>>(yl, outp_w, xo);
    k_head<<<(BB*PRED_LEN*C_OUT + 255)/256, 256, 0, stream>>>(xo, head_w, head_b, stdv, mean, out);
}
